// Decode_Cora_91010357002485
// MI455X (gfx1250) — matrix operand order repaired and checked by reading, not compiled, not run on hardware
//
#include <hip/hip_runtime.h>
#include <hip/hip_fp16.h>

typedef __attribute__((ext_vector_type(16))) _Float16 v16h;
typedef __attribute__((ext_vector_type(8)))  _Float16 v8h;
typedef __attribute__((ext_vector_type(4)))  _Float16 v4h;
typedef __attribute__((ext_vector_type(8)))  float    v8f;
typedef __attribute__((ext_vector_type(4)))  int      v4i;

#define N_NODES 4096
#define F_IN    1433
#define K_FULL  44          // 44 full 32-wide K steps
#define K_TAIL  25          // 1433 - 44*32
#define HC      8
#define DHC     8
#define CDIM    64          // H*DH
#define NEG_SLOPE 0.2f

// ---------------------------------------------------------------------------
// Kernel 1: g = vert @ W   (fp32 inputs, f16 WMMA, fp32 accumulate)
// Block = 128 threads (4 waves); rows [row0,row0+16) x all 64 cols.
// LDS tiles are stored in fragment-native order:
//   As [16][40]  : row m contiguous in k (pad 40 halves = 80B, keeps 16B align)
//   Bt [64][40]  : column c contiguous in k (transposed W tile)
// so each lane builds its WMMA fragments with 2+2 ds_load_b128.
// ---------------------------------------------------------------------------
__global__ __launch_bounds__(128) void gat_gemm_wmma(
    const float* __restrict__ vert, const float* __restrict__ W,
    float* __restrict__ g) {
  __shared__ __align__(16) _Float16 As[16][40];
  __shared__ __align__(16) _Float16 Bt[64][40];

  const int tid  = threadIdx.x;
  const int wave = tid >> 5;      // 0..3
  const int lane = tid & 31;
  const int row0 = blockIdx.x * 16;
  const int col0 = wave * 16;
  const int m    = lane & 15;     // A row / B-C column index
  const int hi   = lane >> 4;     // lane-half selector

  // staging coordinates (fixed per thread)
  const int ar = tid >> 3;              // A row 0..15
  const int ak = (tid & 7) * 4;         // A k group 0,4,..,28
  const int bc = tid & 63;              // B column 0..63
  const int bk = (tid >> 6) * 16;       // B k half: 0 or 16

  const float* aSrc = vert + (size_t)(row0 + ar) * F_IN + ak;
  const float* bSrc = W + bc;

  v8f acc = {};
  for (int ks = 0; ks < K_FULL; ++ks) {
    const int kb = ks * 32;
    __syncthreads();
    // ---- stage A: 4 coalesced b32 loads -> packed b64 store
    {
      float x0 = aSrc[kb + 0], x1 = aSrc[kb + 1], x2 = aSrc[kb + 2], x3 = aSrc[kb + 3];
      v4h p = {(_Float16)x0, (_Float16)x1, (_Float16)x2, (_Float16)x3};
      *(v4h*)&As[ar][ak] = p;
    }
    // ---- stage B transposed: 16 coalesced b32 loads -> two b128 stores
    {
      float xb[16];
#pragma unroll
      for (int kk = 0; kk < 16; ++kk)
        xb[kk] = bSrc[(size_t)(kb + bk + kk) * CDIM];
      v8h p0, p1;
#pragma unroll
      for (int q = 0; q < 8; ++q) { p0[q] = (_Float16)xb[q]; p1[q] = (_Float16)xb[8 + q]; }
      *(v8h*)&Bt[bc][bk]     = p0;
      *(v8h*)&Bt[bc][bk + 8] = p1;
    }
    __syncthreads();
    // ---- build fragments: contiguous 16B slices (ISA 7.12.2 wave32 layout)
    v8h alo = *(const v8h*)&As[m][hi * 8];
    v8h ahi = *(const v8h*)&As[m][16 + hi * 8];
    v8h blo = *(const v8h*)&Bt[col0 + m][hi * 8];
    v8h bhi = *(const v8h*)&Bt[col0 + m][16 + hi * 8];
    v16h a, b;
#pragma unroll
    for (int q = 0; q < 8; ++q) {
      a[q] = alo[q]; a[8 + q] = ahi[q];
      b[q] = blo[q]; b[8 + q] = bhi[q];
    }
    acc = __builtin_amdgcn_wmma_f32_16x16x32_f16(
        false, a, false, b, (short)0, acc, false, false);
  }

  // ---- guarded tail step (k = 1408..1432, 25 valid)
  {
    const int kb = K_FULL * 32;
    __syncthreads();
    {
      v4h p;
#pragma unroll
      for (int q = 0; q < 4; ++q)
        p[q] = (_Float16)((ak + q < K_TAIL) ? aSrc[kb + q] : 0.0f);
      *(v4h*)&As[ar][ak] = p;
    }
    {
      v8h p0, p1;
#pragma unroll
      for (int q = 0; q < 8; ++q) {
        p0[q] = (_Float16)((bk + q < K_TAIL) ? bSrc[(size_t)(kb + bk + q) * CDIM] : 0.0f);
        p1[q] = (_Float16)((bk + 8 + q < K_TAIL) ? bSrc[(size_t)(kb + bk + 8 + q) * CDIM] : 0.0f);
      }
      *(v8h*)&Bt[bc][bk]     = p0;
      *(v8h*)&Bt[bc][bk + 8] = p1;
    }
    __syncthreads();
    v8h alo = *(const v8h*)&As[m][hi * 8];
    v8h ahi = *(const v8h*)&As[m][16 + hi * 8];
    v8h blo = *(const v8h*)&Bt[col0 + m][hi * 8];
    v8h bhi = *(const v8h*)&Bt[col0 + m][16 + hi * 8];
    v16h a, b;
#pragma unroll
    for (int q = 0; q < 8; ++q) {
      a[q] = alo[q]; a[8 + q] = ahi[q];
      b[q] = blo[q]; b[8 + q] = bhi[q];
    }
    acc = __builtin_amdgcn_wmma_f32_16x16x32_f16(
        false, a, false, b, (short)0, acc, false, false);
  }

  // ---- store C: VGPR r -> (M = r + 8*hi, N = lane&15)
  const int mofs = hi ? 8 : 0;
#pragma unroll
  for (int r = 0; r < 8; ++r)
    g[(size_t)(row0 + mofs + r) * CDIM + col0 + m] = acc[r];
}

// ---------------------------------------------------------------------------
// Kernel 2: per-node per-head attention scores sl, sr  (tiny)
// ---------------------------------------------------------------------------
__global__ __launch_bounds__(256) void gat_scores(
    const float* __restrict__ g, const float* __restrict__ a_l,
    const float* __restrict__ a_r, float* __restrict__ sl,
    float* __restrict__ sr) {
  int t = blockIdx.x * blockDim.x + threadIdx.x;  // n*8 + h
  if (t >= N_NODES * HC) return;
  int n = t >> 3, h = t & 7;
  float al = 0.f, ar = 0.f;
#pragma unroll
  for (int d = 0; d < DHC; ++d) {
    float gv = g[(size_t)n * CDIM + h * DHC + d];
    al += gv * a_l[h * DHC + d];
    ar += gv * a_r[h * DHC + d];
  }
  sl[t] = al;
  sr[t] = ar;
}

// ---------------------------------------------------------------------------
// Kernel 3: fused leaky-relu + mask + softmax(axis=j) + aggregation + ELU.
// Block = 256 threads (8 waves), 16 output rows per block.
// Lane owns (i = lane&15, head h = 2*(wave&3) + lane>>4) and 8 channels;
// waves 0-3 handle jj 0..31 of each 64-j chunk, waves 4-7 handle jj 32..63.
// edge rows are folded into a per-lane 32-bit mask register; sr preloaded
// via float4 from a transposed LDS tile. Partials combined through LDS.
// ---------------------------------------------------------------------------
__global__ __launch_bounds__(256) void gat_attn(
    const float* __restrict__ g, const int* __restrict__ edge,
    const float* __restrict__ sl, const float* __restrict__ sr,
    float* __restrict__ out) {
  __shared__ __align__(16) float gS[64][64];   // g chunk   [jj][c]   16 KB
  __shared__ __align__(16) float srT[8][64];   // sr chunk  [h][jj]    2 KB
  __shared__ __align__(16) int   edS[16][64];  // edge rows [i][jj]    4 KB
  __shared__ __align__(16) float slS[16][8];
  __shared__ __align__(16) float accP[16][64]; // partials from waves 4..7
  __shared__ __align__(16) float denP[16][8];

  const int tid  = threadIdx.x;
  const int wave = tid >> 5;
  const int lane = tid & 31;
  const int i0   = blockIdx.x * 16;
  const int iloc = lane & 15;
  const int half = lane >> 4;
  const int w4   = wave & 3;
  const int h    = 2 * w4 + half;      // head 0..7
  const int c0   = h * DHC;            // base output column

  if (tid < 16 * HC)
    slS[tid >> 3][tid & 7] = sl[(size_t)(i0 + (tid >> 3)) * HC + (tid & 7)];
  __syncthreads();
  const float slv = slS[iloc][h];

  float4 acc0 = {0.f, 0.f, 0.f, 0.f}, acc1 = {0.f, 0.f, 0.f, 0.f};
  float den = 0.f;
  const int jb = (wave < 4) ? 0 : 32;

  for (int j0 = 0; j0 < N_NODES; j0 += 64) {
    __syncthreads();
    // ---- stage g chunk: 4 x float4 per thread, coalesced b128
    {
      const int r = tid >> 4, c4 = (tid & 15) * 4;
#pragma unroll
      for (int it = 0; it < 4; ++it)
        *(float4*)&gS[r + it * 16][c4] =
            *(const float4*)&g[(size_t)(j0 + r + it * 16) * CDIM + c4];
    }
    // ---- stage sr transposed: 2 scattered b32 stores per thread
#pragma unroll
    for (int q = 0; q < 2; ++q) {
      int idx = tid * 2 + q;                 // 0..511
      srT[idx & 7][idx >> 3] = sr[(size_t)j0 * HC + idx];
    }
    // ---- stage edge rows: 1 x b128 per thread, non-temporal (single-use)
    {
      const int r = tid >> 4, c4 = (tid & 15) * 4;
      v4i ev = __builtin_nontemporal_load(
          (const v4i*)&edge[(size_t)(i0 + r) * N_NODES + j0 + c4]);
      *(v4i*)&edS[r][c4] = ev;
    }
    if (j0 + 64 < N_NODES && tid < 16)
      __builtin_prefetch(&edge[(size_t)(i0 + tid) * N_NODES + j0 + 64], 0, 0);
    __syncthreads();

    // ---- fold this lane's 32 edge flags into a mask register (8 x b128)
    unsigned emask = 0u;
    const v4i* erow = (const v4i*)&edS[iloc][jb];
#pragma unroll
    for (int q = 0; q < 8; ++q) {
      v4i ev = erow[q];
      emask |= (ev.x ? 1u : 0u) << (4 * q + 0);
      emask |= (ev.y ? 1u : 0u) << (4 * q + 1);
      emask |= (ev.z ? 1u : 0u) << (4 * q + 2);
      emask |= (ev.w ? 1u : 0u) << (4 * q + 3);
    }
    // ---- preload this lane's 32 sr values (8 x b128)
    float srv[32];
#pragma unroll
    for (int q = 0; q < 8; ++q) {
      float4 s4 = *(const float4*)&srT[h][jb + 4 * q];
      srv[4 * q + 0] = s4.x; srv[4 * q + 1] = s4.y;
      srv[4 * q + 2] = s4.z; srv[4 * q + 3] = s4.w;
    }
    // ---- inner: 1 exp + select + 8 FMA per j
#pragma unroll
    for (int jj = 0; jj < 32; ++jj) {
      float e = slv + srv[jj];
      e = (e >= 0.f) ? e : NEG_SLOPE * e;
      float w = ((emask >> jj) & 1u) ? __expf(e) : 0.f;
      den += w;
      const float4* gp = (const float4*)&gS[jb + jj][c0];
      float4 ga = gp[0], gb = gp[1];
      acc0.x += w * ga.x; acc0.y += w * ga.y; acc0.z += w * ga.z; acc0.w += w * ga.w;
      acc1.x += w * gb.x; acc1.y += w * gb.y; acc1.z += w * gb.z; acc1.w += w * gb.w;
    }
  }

  __syncthreads();
  if (wave >= 4) {               // publish partials
    float* ap = &accP[iloc][c0];
    ap[0] = acc0.x; ap[1] = acc0.y; ap[2] = acc0.z; ap[3] = acc0.w;
    ap[4] = acc1.x; ap[5] = acc1.y; ap[6] = acc1.z; ap[7] = acc1.w;
    denP[iloc][h] = den;
  }
  __syncthreads();
  if (wave < 4) {                // combine, normalize, ELU, store
    const float* ap = &accP[iloc][c0];
    float d0 = den + denP[iloc][h];
    float inv = 1.0f / d0;
    float o[8] = {acc0.x + ap[0], acc0.y + ap[1], acc0.z + ap[2], acc0.w + ap[3],
                  acc1.x + ap[4], acc1.y + ap[5], acc1.z + ap[6], acc1.w + ap[7]};
    float* op = &out[(size_t)(i0 + iloc) * CDIM + c0];
#pragma unroll
    for (int d = 0; d < 8; ++d) {
      float x = o[d] * inv;
      op[d] = (x > 0.f) ? x : (__expf(x) - 1.0f);
    }
  }
}

// ---------------------------------------------------------------------------
extern "C" void kernel_launch(void* const* d_in, const int* in_sizes, int n_in,
                              void* d_out, int out_size, void* d_ws, size_t ws_size,
                              hipStream_t stream) {
  (void)in_sizes; (void)n_in; (void)out_size; (void)ws_size;
  const float* vert = (const float*)d_in[0];
  const int*   edge = (const int*)d_in[1];
  const float* W    = (const float*)d_in[2];
  const float* a_l  = (const float*)d_in[3];
  const float* a_r  = (const float*)d_in[4];
  float* out = (float*)d_out;

  // workspace layout (floats): g [4096*64] | sl [4096*8] | sr [4096*8]
  float* g  = (float*)d_ws;
  float* sl = g + (size_t)N_NODES * CDIM;
  float* sr = sl + (size_t)N_NODES * HC;

  gat_gemm_wmma<<<N_NODES / 16, 128, 0, stream>>>(vert, W, g);
  gat_scores<<<(N_NODES * HC + 255) / 256, 256, 0, stream>>>(g, a_l, a_r, sl, sr);
  gat_attn<<<N_NODES / 16, 256, 0, stream>>>(g, edge, sl, sr, out);
}
